// DeformableTransformer_6854767805320
// MI455X (gfx1250) — compile-verified, matrix operand order checked by reading, not run on hardware
//
#include <hip/hip_runtime.h>

// ---------------------------------------------------------------------------
// Deformable-DETR transformer for MI455X (gfx1250), wave32.
// bf16 operands end-to-end: weights converted once per launch, activation
// producers emit bf16 twins, GEMM = double-buffered async global->LDS staging
// feeding v_wmma_f32_16x16x32_bf16 (pure ds_load_b128 -> wmma inner loop).
// ---------------------------------------------------------------------------

#define LEN_V_C 5440
#define N_BATCH 2
#define LQ_DEC  300

#if defined(__AMDGCN__) && __has_builtin(__builtin_amdgcn_global_load_async_to_lds_b128)
#define DT_ASYNC 1
#else
#define DT_ASYNC 0
#endif

typedef __bf16 bf16;
typedef bf16  bf16x8 __attribute__((ext_vector_type(8)));
typedef bf16  v16bf  __attribute__((ext_vector_type(16)));
typedef float v8f    __attribute__((ext_vector_type(8)));
typedef unsigned int u32;

// f32 -> bf16 round-to-nearest-even
__device__ __forceinline__ bf16 f2bf(float f) {
  union { float f; u32 u; } a; a.f = f;
  u32 r = a.u + 0x7FFFu + ((a.u >> 16) & 1u);
  union { unsigned short s; bf16 b; } o; o.s = (unsigned short)(r >> 16);
  return o.b;
}
// two f32 -> packed dword of two bf16 (RNE) via v_perm_b32
__device__ __forceinline__ u32 pack_bf2(float flo, float fhi) {
  union { float f; u32 u; } a, b; a.f = flo; b.f = fhi;
  u32 lo = a.u + 0x7FFFu + ((a.u >> 16) & 1u);
  u32 hi = b.u + 0x7FFFu + ((b.u >> 16) & 1u);
  return __builtin_amdgcn_perm(hi, lo, 0x07060302u);
}

union fragu { struct { bf16x8 lo, hi; } p; v16bf v; };
// 16x32 bf16 A/B fragment (ISA 7.12.2): runs of 8 halves at K=kq and K=16+kq.
__device__ __forceinline__ v16bf load_frag(const bf16* p, int kq) {
  fragu u;
  u.p.lo = *(const bf16x8*)(p + kq);
  u.p.hi = *(const bf16x8*)(p + 16 + kq);
  return u.v;
}

#if DT_ASYNC
typedef int v4i __attribute__((vector_size(4 * sizeof(int))));
#endif
template <int N>
__device__ __forceinline__ void dt_wait_async() {
#if __has_builtin(__builtin_amdgcn_s_wait_asynccnt)
  __builtin_amdgcn_s_wait_asynccnt(N);
#else
  asm volatile("s_wait_asynccnt %0" :: "i"(N) : "memory");
#endif
}

// ---------------------------------------------------------------------------
// GEMM:  {C,Cbf}[M,N] = act( A[M,K](bf16) @ W[N,K]^T(bf16) + bias[N] )
// 128x64 tile / 128-thread block; wave w owns rows [32w,32w+32):
// 2 A-frags x 4 B-frags -> 8 WMMA per wave per K-step. bf16 LDS tiles,
// 40-half padded stride (conflict-free per-half-wave b128 reads).
// Double-buffered staging: 6 async b128 per thread per tile; ASYNCcnt is
// in-order so s_wait_asynccnt 6 retires the previous tile.
// ---------------------------------------------------------------------------
#define TM 128
#define TN 64
#define TK 32
#define LDK 40

__global__ __launch_bounds__(128)
void dt_gemm_bf16(const bf16* __restrict__ A, const bf16* __restrict__ W,
                  const float* __restrict__ bias, float* __restrict__ C,
                  bf16* __restrict__ Cbf, int M, int N, int K, int relu)
{
  __shared__ __align__(16) bf16 lA[2][TM * LDK];
  __shared__ __align__(16) bf16 lB[2][TN * LDK];

  const int m0 = blockIdx.x * TM;
  const int n0 = blockIdx.y * TN;
  const int tid = threadIdx.x;
  const int wave = tid >> 5;
  const int lane = tid & 31;

  v8f acc[2][4] = {};

  auto stage = [&](int k0, int buf) {
#pragma unroll
    for (int it = 0; it < 6; ++it) {
      int c = tid + it * 128;            // wave-uniform A/B split per it
      int row = c >> 2;                  // 0..191 (4 b128 chunks per 32-K row)
      int kc  = (c & 3) << 3;            // 0,8,16,24 halves
      const bf16* gp;
      bf16* lp;
      if (row < TM) {
        int gr = m0 + row; if (gr > M - 1) gr = M - 1;   // clamp: never stored
        gp = A + (size_t)gr * K + k0 + kc;
        lp = &lA[buf][row * LDK + kc];
      } else {
        gp = W + (size_t)(n0 + row - TM) * K + k0 + kc;  // N % 64 == 0
        lp = &lB[buf][(row - TM) * LDK + kc];
      }
#if DT_ASYNC
      __builtin_amdgcn_global_load_async_to_lds_b128((v4i*)gp, (v4i*)lp, 0, 0);
#else
      *(float4*)lp = *(const float4*)gp;
#endif
    }
  };

  const int NT = K / TK;
#if DT_ASYNC
  stage(0, 0);
#endif
  for (int t = 0; t < NT; ++t) {
#if DT_ASYNC
    if (t + 1 < NT) { stage((t + 1) * TK, (t + 1) & 1); dt_wait_async<6>(); }
    else            { dt_wait_async<0>(); }
#else
    stage(t * TK, t & 1);
#endif
    __syncthreads();

    const int buf = t & 1;
    const int r  = lane & 15;
    const int kq = (lane >> 4) << 3;     // 0 or 8
    const bf16* pa = &lA[buf][(wave * 32 + r) * LDK];
    v16bf a0 = load_frag(pa, kq);
    v16bf a1 = load_frag(pa + 16 * LDK, kq);
    v16bf b0 = load_frag(&lB[buf][(r     ) * LDK], kq);
    v16bf b1 = load_frag(&lB[buf][(r + 16) * LDK], kq);
    v16bf b2 = load_frag(&lB[buf][(r + 32) * LDK], kq);
    v16bf b3 = load_frag(&lB[buf][(r + 48) * LDK], kq);

    acc[0][0] = __builtin_amdgcn_wmma_f32_16x16x32_bf16(false, a0, false, b0, (short)0, acc[0][0], false, false);
    acc[0][1] = __builtin_amdgcn_wmma_f32_16x16x32_bf16(false, a0, false, b1, (short)0, acc[0][1], false, false);
    acc[0][2] = __builtin_amdgcn_wmma_f32_16x16x32_bf16(false, a0, false, b2, (short)0, acc[0][2], false, false);
    acc[0][3] = __builtin_amdgcn_wmma_f32_16x16x32_bf16(false, a0, false, b3, (short)0, acc[0][3], false, false);
    acc[1][0] = __builtin_amdgcn_wmma_f32_16x16x32_bf16(false, a1, false, b0, (short)0, acc[1][0], false, false);
    acc[1][1] = __builtin_amdgcn_wmma_f32_16x16x32_bf16(false, a1, false, b1, (short)0, acc[1][1], false, false);
    acc[1][2] = __builtin_amdgcn_wmma_f32_16x16x32_bf16(false, a1, false, b2, (short)0, acc[1][2], false, false);
    acc[1][3] = __builtin_amdgcn_wmma_f32_16x16x32_bf16(false, a1, false, b3, (short)0, acc[1][3], false, false);
    __syncthreads();
  }

  // C/D layout: VGPR i -> M = i + 8*(lane>>4), N = lane&15
  const int cn = lane & 15;
  const int half = lane >> 4;
#pragma unroll
  for (int ai = 0; ai < 2; ++ai)
#pragma unroll
    for (int j = 0; j < 4; ++j) {
      int gn = n0 + j * 16 + cn;
      float bv = bias ? bias[gn] : 0.f;
#pragma unroll
      for (int i = 0; i < 8; ++i) {
        int gm = m0 + wave * 32 + ai * 16 + half * 8 + i;
        if (gm < M) {
          float v = acc[ai][j][i] + bv;
          if (relu) v = v > 0.f ? v : 0.f;
          size_t idx = (size_t)gm * N + gn;
          if (C)   C[idx] = v;
          if (Cbf) Cbf[idx] = f2bf(v);
        }
      }
    }
}

// ---------------------------------------------------------------------------
// Conversion / elementwise helpers
// ---------------------------------------------------------------------------
__global__ void dt_f2bf(const float* __restrict__ a, u32* __restrict__ b, int n2) {
  int i = blockIdx.x * blockDim.x + threadIdx.x;    // over n/2 packed pairs
  if (i < n2) b[i] = pack_bf2(a[2 * i], a[2 * i + 1]);
}

__global__ void dt_cpbf(const float* __restrict__ a, float* __restrict__ f,
                        bf16* __restrict__ b, int n) {
  int i = blockIdx.x * blockDim.x + threadIdx.x;
  if (i < n) { float v = a[i]; f[i] = v; b[i] = f2bf(v); }
}

__global__ void dt_add_bf(const float* __restrict__ a, const float* __restrict__ b,
                          bf16* __restrict__ c, int n) {
  int i = blockIdx.x * blockDim.x + threadIdx.x;
  if (i < n) c[i] = f2bf(a[i] + b[i]);
}

__global__ void dt_copy(const float* __restrict__ a, float* __restrict__ b, int n) {
  int i = blockIdx.x * blockDim.x + threadIdx.x;
  if (i < n) b[i] = a[i];
}

// ---------------------------------------------------------------------------
// Fused residual-add + LayerNorm over D=256; emits f32 + optional bf16 twin.
// ---------------------------------------------------------------------------
__device__ __forceinline__ float block_sum256(float v, float* red) {
  for (int o = 16; o > 0; o >>= 1) v += __shfl_down(v, o, 32);
  int lane = threadIdx.x & 31, w = threadIdx.x >> 5;
  if (lane == 0) red[w] = v;
  __syncthreads();
  if (w == 0) {
    float t = (lane < 8) ? red[lane] : 0.f;
    for (int o = 4; o > 0; o >>= 1) t += __shfl_down(t, o, 32);
    if (lane == 0) red[0] = t;
  }
  __syncthreads();
  float r = red[0];
  __syncthreads();
  return r;
}

__global__ __launch_bounds__(256)
void dt_add_ln(const float* __restrict__ X, const float* __restrict__ Y,
               const float* __restrict__ G, const float* __restrict__ B,
               float* __restrict__ O, bf16* __restrict__ Obf)
{
  __shared__ float red[8];
  int row = blockIdx.x, t = threadIdx.x;
  size_t base = (size_t)row * 256 + t;
  float v = X[base] + (Y ? Y[base] : 0.f);
  float mean = block_sum256(v, red) * (1.f / 256.f);
  float u = v - mean;
  float var = block_sum256(u * u, red) * (1.f / 256.f);
  float rs = rsqrtf(var + 1e-5f);
  float o = u * rs * G[t] + B[t];
  O[base] = o;
  if (Obf) Obf[base] = f2bf(o);
}

// ---------------------------------------------------------------------------
// Multi-scale deformable attention core. One 256-thread block per (n,q);
// bilinear-gather from f32 value tensor (N, LEN_V, 8, 32); bf16 output.
// ---------------------------------------------------------------------------
__global__ __launch_bounds__(256)
void dt_msda(const float* __restrict__ V, const float* __restrict__ OFF,
             const float* __restrict__ AWL, const float* __restrict__ REF,
             bf16* __restrict__ OUT, int Lq)
{
  const int Hs[4] = {64, 32, 16, 8};
  const int Ws[4] = {64, 32, 16, 8};
  const int Ss[4] = {0, 4096, 5120, 5376};

  int nq = blockIdx.x;
  int n = nq / Lq;
  int t = threadIdx.x;
  int h = t >> 5, d = t & 31;

  const float* logit = AWL + (size_t)nq * 128 + h * 16;
  float w[16], mx = -1e30f;
#pragma unroll
  for (int i = 0; i < 16; ++i) { w[i] = logit[i]; mx = fmaxf(mx, w[i]); }
  float s = 0.f;
#pragma unroll
  for (int i = 0; i < 16; ++i) { w[i] = __expf(w[i] - mx); s += w[i]; }
  float inv = 1.f / s;

  const float* off = OFF + (size_t)nq * 256 + h * 32;
  const float* ref = REF + (size_t)nq * 8;
  float acc = 0.f;

#pragma unroll
  for (int l = 0; l < 4; ++l) {
    int Hl = Hs[l], Wl = Ws[l];
    const float* vb = V + ((size_t)n * LEN_V_C + Ss[l]) * 256 + h * 32 + d;
    float rx = ref[l * 2 + 0], ry = ref[l * 2 + 1];
#pragma unroll
    for (int p = 0; p < 4; ++p) {
      float ox = off[(l * 4 + p) * 2 + 0];
      float oy = off[(l * 4 + p) * 2 + 1];
      float x = rx * Wl + ox - 0.5f;     // (ref + off/norm)*W - 0.5
      float y = ry * Hl + oy - 0.5f;
      float fx = floorf(x), fy = floorf(y);
      float lx = x - fx, ly = y - fy;
      int x0 = (int)fx, y0 = (int)fy, x1 = x0 + 1, y1 = y0 + 1;
      int cx0 = x0 < 0 ? 0 : (x0 > Wl - 1 ? Wl - 1 : x0);
      int cx1 = x1 < 0 ? 0 : (x1 > Wl - 1 ? Wl - 1 : x1);
      int cy0 = y0 < 0 ? 0 : (y0 > Hl - 1 ? Hl - 1 : y0);
      int cy1 = y1 < 0 ? 0 : (y1 > Hl - 1 ? Hl - 1 : y1);
      bool vx0 = (x0 >= 0) & (x0 < Wl), vx1 = (x1 >= 0) & (x1 < Wl);
      bool vy0 = (y0 >= 0) & (y0 < Hl), vy1 = (y1 >= 0) & (y1 < Hl);
      float v00 = (vx0 && vy0) ? vb[(size_t)(cy0 * Wl + cx0) * 256] : 0.f;
      float v10 = (vx1 && vy0) ? vb[(size_t)(cy0 * Wl + cx1) * 256] : 0.f;
      float v01 = (vx0 && vy1) ? vb[(size_t)(cy1 * Wl + cx0) * 256] : 0.f;
      float v11 = (vx1 && vy1) ? vb[(size_t)(cy1 * Wl + cx1) * 256] : 0.f;
      float smp = v00 * (1 - lx) * (1 - ly) + v10 * lx * (1 - ly)
                + v01 * (1 - lx) * ly       + v11 * lx * ly;
      acc += w[l * 4 + p] * inv * smp;
    }
  }
  OUT[(size_t)nq * 256 + h * 32 + d] = f2bf(acc);
}

// ---------------------------------------------------------------------------
// Decoder self-attention: one 64-thread block per (n, head, q); bf16 output.
// ---------------------------------------------------------------------------
__global__ __launch_bounds__(64)
void dt_attn(const float* __restrict__ QK, const float* __restrict__ VD,
             bf16* __restrict__ O, int L)
{
  int b = blockIdx.x;
  int q = b % L; b /= L;
  int h = b % 8; int n = b / 8;
  int t = threadIdx.x;
  __shared__ float qv[32];
  __shared__ float p[LQ_DEC];
  __shared__ float red[64];

  const float* qrow = QK + (size_t)(n * L + q) * 512 + h * 32;
  if (t < 32) qv[t] = qrow[t] * 0.1767766953f;   // 32^-0.5
  __syncthreads();

  float mx = -1e30f;
  for (int k = t; k < L; k += 64) {
    const float* krow = QK + (size_t)(n * L + k) * 512 + 256 + h * 32;
    float s = 0.f;
#pragma unroll
    for (int d2 = 0; d2 < 32; ++d2) s += qv[d2] * krow[d2];
    p[k] = s; mx = fmaxf(mx, s);
  }
  red[t] = mx; __syncthreads();
  for (int o = 32; o > 0; o >>= 1) { if (t < o) red[t] = fmaxf(red[t], red[t + o]); __syncthreads(); }
  mx = red[0]; __syncthreads();

  float sum = 0.f;
  for (int k = t; k < L; k += 64) { float e = __expf(p[k] - mx); p[k] = e; sum += e; }
  red[t] = sum; __syncthreads();
  for (int o = 32; o > 0; o >>= 1) { if (t < o) red[t] += red[t + o]; __syncthreads(); }
  float inv = 1.f / red[0]; __syncthreads();

  int d = t & 31, half = t >> 5;
  float acc = 0.f;
  for (int k = half; k < L; k += 2)
    acc += p[k] * VD[(size_t)(n * L + k) * 256 + h * 32 + d];
  red[t] = acc; __syncthreads();
  if (t < 32) O[(size_t)(n * L + q) * 256 + h * 32 + t] = f2bf((red[t] + red[t + 32]) * inv);
}

// ---------------------------------------------------------------------------
// Reference-point helpers / bbox head tail
// ---------------------------------------------------------------------------
__global__ void dt_enc_ref(const float* __restrict__ vr, float* __restrict__ ref) {
  int i = blockIdx.x * blockDim.x + threadIdx.x;      // over 2*LEN_V*4
  if (i >= N_BATCH * LEN_V_C * 4) return;
  int l = i & 3, q = (i >> 2) % LEN_V_C, n = i / (LEN_V_C * 4);
  int lvl = (q < 4096) ? 0 : (q < 5120) ? 1 : (q < 5376) ? 2 : 3;
  const int Hs[4] = {64, 32, 16, 8}, Ws[4] = {64, 32, 16, 8}, Ss[4] = {0, 4096, 5120, 5376};
  int Hl = Hs[lvl], Wl = Ws[lvl];
  int rq = q - Ss[lvl];
  int r = rq / Wl, c = rq % Wl;
  float rx = (c + 0.5f) / (vr[(n * 4 + lvl) * 2 + 0] * Wl);
  float ry = (r + 0.5f) / (vr[(n * 4 + lvl) * 2 + 1] * Hl);
  ref[(size_t)i * 2 + 0] = rx * vr[(n * 4 + l) * 2 + 0];
  ref[(size_t)i * 2 + 1] = ry * vr[(n * 4 + l) * 2 + 1];
}

__global__ void dt_dec_ref(const float* __restrict__ ref, const float* __restrict__ vr,
                           float* __restrict__ out) {
  int i = blockIdx.x * blockDim.x + threadIdx.x;      // over 2*300*4*2
  if (i >= N_BATCH * LQ_DEC * 8) return;
  int c = i & 1, l = (i >> 1) & 3, q = (i >> 3) % LQ_DEC, n = i / (LQ_DEC * 8);
  out[i] = ref[(size_t)(n * LQ_DEC + q) * 2 + c] * vr[(n * 4 + l) * 2 + c];
}

__global__ __launch_bounds__(64)
void dt_linear2(const float* __restrict__ X, const float* __restrict__ W2,
                const float* __restrict__ b2, float* __restrict__ Y) {
  int row = blockIdx.x, t = threadIdx.x;
  __shared__ float r0[64], r1[64];
  float a0 = 0.f, a1 = 0.f;
  for (int k = t; k < 256; k += 64) {
    float x = X[(size_t)row * 256 + k];
    a0 += x * W2[k]; a1 += x * W2[256 + k];
  }
  r0[t] = a0; r1[t] = a1; __syncthreads();
  for (int o = 32; o > 0; o >>= 1) { if (t < o) { r0[t] += r0[t + o]; r1[t] += r1[t + o]; } __syncthreads(); }
  if (t == 0) { Y[row * 2] = r0[0] + b2[0]; Y[row * 2 + 1] = r1[0] + b2[1]; }
}

__global__ void dt_ref_update(const float* __restrict__ T, float* __restrict__ R, int n) {
  int i = blockIdx.x * blockDim.x + threadIdx.x;
  if (i >= n) return;
  float r = fminf(fmaxf(R[i], 0.f), 1.f);
  float inv_sig = __logf(fmaxf(r, 1e-5f) / fmaxf(1.f - r, 1e-5f));
  float t = T[i] + inv_sig;
  R[i] = 1.f / (1.f + __expf(-t));
}

// ---------------------------------------------------------------------------
// Host orchestration
// ---------------------------------------------------------------------------
extern "C" void kernel_launch(void* const* d_in, const int* in_sizes, int n_in,
                              void* d_out, int out_size, void* d_ws, size_t ws_size,
                              hipStream_t stream)
{
  (void)in_sizes; (void)n_in; (void)out_size; (void)ws_size;

  const float* src  = (const float*)d_in[0];
  const float* pos  = (const float*)d_in[1];
  const float* vr   = (const float*)d_in[2];
  const float* tgt  = (const float*)d_in[3];
  const float* qpos = (const float*)d_in[4];
  const float* rpts = (const float*)d_in[5];
  // d_in[6] spatial_shapes, d_in[7] level_start_index: compile-time constants here.

  const float* e_so_w = (const float*)d_in[8];   const float* e_so_b = (const float*)d_in[9];
  const float* e_aw_w = (const float*)d_in[10];  const float* e_aw_b = (const float*)d_in[11];
  const float* e_vp_w = (const float*)d_in[12];  const float* e_vp_b = (const float*)d_in[13];
  const float* e_op_w = (const float*)d_in[14];  const float* e_op_b = (const float*)d_in[15];
  const float* e_l1_w = (const float*)d_in[16];  const float* e_l1_b = (const float*)d_in[17];
  const float* e_l2_w = (const float*)d_in[18];  const float* e_l2_b = (const float*)d_in[19];
  const float* e_n1g  = (const float*)d_in[20];  const float* e_n1b  = (const float*)d_in[21];
  const float* e_n2g  = (const float*)d_in[22];  const float* e_n2b  = (const float*)d_in[23];

  const float* d_so_w = (const float*)d_in[24];  const float* d_so_b = (const float*)d_in[25];
  const float* d_aw_w = (const float*)d_in[26];  const float* d_aw_b = (const float*)d_in[27];
  const float* d_vp_w = (const float*)d_in[28];  const float* d_vp_b = (const float*)d_in[29];
  const float* d_op_w = (const float*)d_in[30];  const float* d_op_b = (const float*)d_in[31];
  const float* d_l1_w = (const float*)d_in[32];  const float* d_l1_b = (const float*)d_in[33];
  const float* d_l2_w = (const float*)d_in[34];  const float* d_l2_b = (const float*)d_in[35];
  const float* d_n1g  = (const float*)d_in[36];  const float* d_n1b  = (const float*)d_in[37];
  const float* d_n2g  = (const float*)d_in[38];  const float* d_n2b  = (const float*)d_in[39];
  const float* d_in_w = (const float*)d_in[40];  const float* d_in_b = (const float*)d_in[41];
  const float* d_ou_w = (const float*)d_in[42];  const float* d_ou_b = (const float*)d_in[43];
  const float* d_n3g  = (const float*)d_in[44];  const float* d_n3b  = (const float*)d_in[45];
  const float* ce0_w  = (const float*)d_in[46];  const float* ce0_b  = (const float*)d_in[47];
  const float* ce1_w  = (const float*)d_in[48];  const float* ce1_b  = (const float*)d_in[49];
  const float* ce2_w  = (const float*)d_in[50];  const float* ce2_b  = (const float*)d_in[51];

  float* out = (float*)d_out;
  float* ws  = (float*)d_ws;

  const int ME  = N_BATCH * LEN_V_C;      // 10880
  const int MD  = N_BATCH * LQ_DEC;       // 600
  const size_t SZE = (size_t)ME * 256;

  // ---- f32 buffers (units of float) ----
  float* X    = ws;                       // encoder activations / memory
  float* O    = X + SZE;
  float* V    = O + SZE;                  // msda value (f32)
  float* SO   = V + SZE;                  // msda sampling offsets
  float* AW   = SO + SZE;                 // msda logits (ME*128)
  float* REFE = AW + SZE / 2;
  float* OUTD = REFE + (size_t)ME * 8;
  float* REFD = OUTD + (size_t)MD * 256;
  float* REFI = REFD + (size_t)MD * 2;
  float* QK   = REFI + (size_t)MD * 8;
  float* VD   = QK + (size_t)MD * 512;
  float* T1   = VD + (size_t)MD * 256;
  float* T3   = T1 + (size_t)MD * 256;
  float* SOD  = T3 + (size_t)MD * 256;
  float* AWD  = SOD + (size_t)MD * 256;
  float* TCE  = AWD + (size_t)MD * 128;
  float* fend = TCE + (size_t)MD * 2;

  // ---- bf16 buffers (2 per float slot) ----
  bf16* Xbf    = (bf16*)fend;
  bf16* XPbf   = Xbf + SZE;               // x + pos
  bf16* Sbf    = XPbf + SZE;              // msda output
  bf16* Hbf    = Sbf + SZE;               // FFN hidden (ME*1024)
  bf16* OUTDbf = Hbf + (size_t)ME * 1024;
  bf16* OUTQbf = OUTDbf + (size_t)MD * 256;
  bf16* AObf   = OUTQbf + (size_t)MD * 256;
  bf16* SDbf   = AObf + (size_t)MD * 256;
  bf16* T2bf   = SDbf + (size_t)MD * 256;
  bf16* HDbf   = T2bf + (size_t)MD * 256;
  u32*  WB     = (u32*)(HDbf + (size_t)MD * 1024);   // bf16 weight arena

  auto gemm = [&](const bf16* Ap, const bf16* Wp, const float* bp,
                  float* Cp, bf16* Cbfp, int Mm, int Nn, int Kk, int relu) {
    dim3 g((Mm + TM - 1) / TM, Nn / TN);
    dt_gemm_bf16<<<g, 128, 0, stream>>>(Ap, Wp, bp, Cp, Cbfp, Mm, Nn, Kk, relu);
  };
  auto copyf = [&](const float* a, float* b, int n) {
    dt_copy<<<(n + 255) / 256, 256, 0, stream>>>(a, b, n);
  };
  auto addbf = [&](const float* a, const float* b, bf16* c, int n) {
    dt_add_bf<<<(n + 255) / 256, 256, 0, stream>>>(a, b, c, n);
  };

  // ---- convert all GEMM weights to bf16 once per launch ----
  size_t wb_off = 0;
  auto cvt = [&](const float* w, size_t n) -> const bf16* {
    u32* dst = WB + wb_off;
    int n2 = (int)(n / 2);
    dt_f2bf<<<(n2 + 255) / 256, 256, 0, stream>>>(w, dst, n2);
    wb_off += n / 2;
    return (const bf16*)dst;
  };
  const size_t W256 = 6 * 256 * 256, W128 = 6 * 128 * 256;
  const size_t W1K  = 6 * 1024 * 256, W768 = 6 * 768 * 256;
  const bf16* esoW = cvt(e_so_w, W256); const bf16* eawW = cvt(e_aw_w, W128);
  const bf16* evpW = cvt(e_vp_w, W256); const bf16* eopW = cvt(e_op_w, W256);
  const bf16* el1W = cvt(e_l1_w, W1K);  const bf16* el2W = cvt(e_l2_w, W1K);
  const bf16* dsoW = cvt(d_so_w, W256); const bf16* dawW = cvt(d_aw_w, W128);
  const bf16* dvpW = cvt(d_vp_w, W256); const bf16* dopW = cvt(d_op_w, W256);
  const bf16* dl1W = cvt(d_l1_w, W1K);  const bf16* dl2W = cvt(d_l2_w, W1K);
  const bf16* dinW = cvt(d_in_w, W768); const bf16* douW = cvt(d_ou_w, W256);
  const bf16* c0W  = cvt(ce0_w,  W256); const bf16* c1W  = cvt(ce1_w,  W256);

  // ---------------- Encoder ----------------
  dt_cpbf<<<((int)SZE + 255) / 256, 256, 0, stream>>>(src, X, Xbf, (int)SZE);
  dt_enc_ref<<<(N_BATCH * LEN_V_C * 4 + 255) / 256, 256, 0, stream>>>(vr, REFE);

  for (int l = 0; l < 6; ++l) {
    size_t o256 = (size_t)l * 256 * 256, o128 = (size_t)l * 128 * 256;
    size_t o1k  = (size_t)l * 1024 * 256;
    gemm(Xbf, evpW + o256, e_vp_b + l * 256, V, nullptr, ME, 256, 256, 0);
    addbf(X, pos, XPbf, (int)SZE);
    gemm(XPbf, esoW + o256, e_so_b + l * 256, SO, nullptr, ME, 256, 256, 0);
    gemm(XPbf, eawW + o128, e_aw_b + l * 128, AW, nullptr, ME, 128, 256, 0);
    dt_msda<<<ME, 256, 0, stream>>>(V, SO, AW, REFE, Sbf, LEN_V_C);
    gemm(Sbf, eopW + o256, e_op_b + l * 256, O, nullptr, ME, 256, 256, 0);
    dt_add_ln<<<ME, 256, 0, stream>>>(X, O, e_n1g + l * 256, e_n1b + l * 256, X, Xbf);
    gemm(Xbf, el1W + o1k, e_l1_b + l * 1024, nullptr, Hbf, ME, 1024, 256, 1);
    gemm(Hbf, el2W + o1k, e_l2_b + l * 256,  O, nullptr, ME, 256, 1024, 0);
    dt_add_ln<<<ME, 256, 0, stream>>>(X, O, e_n2g + l * 256, e_n2b + l * 256, X, Xbf);
  }

  // ---------------- Decoder ----------------
  dt_cpbf<<<(MD * 256 + 255) / 256, 256, 0, stream>>>(tgt, OUTD, OUTDbf, MD * 256);
  copyf(rpts, REFD, MD * 2);

  for (int l = 0; l < 6; ++l) {
    size_t o256 = (size_t)l * 256 * 256, o128 = (size_t)l * 128 * 256;
    size_t o1k  = (size_t)l * 1024 * 256, o768 = (size_t)l * 768 * 256;

    dt_dec_ref<<<(MD * 8 + 255) / 256, 256, 0, stream>>>(REFD, vr, REFI);

    // self-attention
    addbf(OUTD, qpos, OUTQbf, MD * 256);
    gemm(OUTQbf, dinW + o768,             d_in_b + l * 768,       QK, nullptr, MD, 512, 256, 0);
    gemm(OUTDbf, dinW + o768 + 512 * 256, d_in_b + l * 768 + 512, VD, nullptr, MD, 256, 256, 0);
    dt_attn<<<N_BATCH * 8 * LQ_DEC, 64, 0, stream>>>(QK, VD, AObf, LQ_DEC);
    gemm(AObf, douW + o256, d_ou_b + l * 256, T1, nullptr, MD, 256, 256, 0);
    dt_add_ln<<<MD, 256, 0, stream>>>(OUTD, T1, d_n2g + l * 256, d_n2b + l * 256, OUTD, OUTDbf);

    // cross deformable attention (value = vp(memory), per-layer weights)
    gemm(Xbf, dvpW + o256, d_vp_b + l * 256, V, nullptr, ME, 256, 256, 0);
    addbf(OUTD, qpos, OUTQbf, MD * 256);          // out changed after LN
    gemm(OUTQbf, dsoW + o256, d_so_b + l * 256, SOD, nullptr, MD, 256, 256, 0);
    gemm(OUTQbf, dawW + o128, d_aw_b + l * 128, AWD, nullptr, MD, 128, 256, 0);
    dt_msda<<<MD, 256, 0, stream>>>(V, SOD, AWD, REFI, SDbf, LQ_DEC);
    gemm(SDbf, dopW + o256, d_op_b + l * 256, T1, nullptr, MD, 256, 256, 0);
    dt_add_ln<<<MD, 256, 0, stream>>>(OUTD, T1, d_n1g + l * 256, d_n1b + l * 256, OUTD, OUTDbf);

    // FFN
    gemm(OUTDbf, dl1W + o1k, d_l1_b + l * 1024, nullptr, HDbf, MD, 1024, 256, 1);
    gemm(HDbf,   dl2W + o1k, d_l2_b + l * 256,  T1, nullptr, MD, 256, 1024, 0);
    dt_add_ln<<<MD, 256, 0, stream>>>(OUTD, T1, d_n3g + l * 256, d_n3b + l * 256, OUTD, OUTDbf);

    // bbox refinement head
    gemm(OUTDbf, c0W + o256, ce0_b + l * 256, nullptr, T2bf, MD, 256, 256, 1);
    gemm(T2bf,   c1W + o256, ce1_b + l * 256, T3, nullptr, MD, 256, 256, 1);
    dt_linear2<<<MD, 64, 0, stream>>>(T3, ce2_w + (size_t)l * 2 * 256, ce2_b + l * 2, TCE);
    dt_ref_update<<<(MD * 2 + 255) / 256, 256, 0, stream>>>(TCE, REFD, MD * 2);
  }

  // outputs: hs (2,300,256) then ref (2,300,2), concatenated flat
  copyf(OUTD, out, MD * 256);
  copyf(REFD, out + (size_t)MD * 256, MD * 2);
}
